// Transformer_25237227831941
// MI455X (gfx1250) — matrix operand order repaired and checked by reading, not compiled, not run on hardware
//
#include <hip/hip_runtime.h>

#define B_ 4
#define S_ 512
#define D_ 512
#define H_ 8
#define DH_ 64
#define F_ 2048
#define V_ 32000
#define NL_ 6
#define MROWS (B_*S_)   // 2048

typedef __attribute__((ext_vector_type(16))) __bf16 v16bf;
typedef __attribute__((ext_vector_type(8)))  float  v8f;

union FragBF { v16bf v; unsigned short u[16]; uint4 q[2]; };

__device__ __forceinline__ unsigned short f2bf(float f) {
  unsigned int u = __float_as_uint(f);
  return (unsigned short)((u + 0x7FFFu + ((u >> 16) & 1u)) >> 16);
}
// pack two f32 -> two bf16 in one dword (lo = a, hi = b)
__device__ __forceinline__ unsigned int f2bf_pk(float a, float b) {
  return (unsigned int)f2bf(a) | ((unsigned int)f2bf(b) << 16);
}

// ---------------------------------------------------------------------------
// GEMM: C[M,N] = act(A[M,K] @ B[K,N] [+ bias]) [+ resid]
// 128x128 tile / 256 threads (8 waves). Each wave owns 32x64.
// K-step 64 => 16 WMMA per wave per barrier. Double-buffered LDS staging:
// next tile's global loads overlap current tile's WMMA; one barrier per step.
// ---------------------------------------------------------------------------
template<bool HAS_BIAS, bool RELU, bool HAS_RES>
__global__ __launch_bounds__(256)
void gemm_bf16(const float* __restrict__ A, const float* __restrict__ Bm,
               const float* __restrict__ bias, const float* __restrict__ resid,
               float* __restrict__ C, int M, int N, int K)
{
  __shared__ __align__(16) unsigned short As[2][128][72];  // [buf][m][k], 144B rows
  __shared__ __align__(16) unsigned short Bs[2][128][72];  // [buf][n][k] (transposed)

  const int tid  = threadIdx.x;
  const int lane = tid & 31;
  const int wave = tid >> 5;
  const int tileM = blockIdx.y * 128, tileN = blockIdx.x * 128;
  const int mrow0 = (wave >> 1) * 32;   // 4 row-groups of 32
  const int ncol0 = (wave & 1) * 64;    // 2 col-groups of 64
  const int khalfA = (lane >> 4) * 8;   // A frag K-offset (shorts)
  const int kb0    = (lane >> 4) * 8;   // B frag K-offset (shorts)

  // staging indices: 128x64 tile = 2048 float4s, 8 per thread
  const int ar = tid >> 4, ac4 = (tid & 15) * 4;   // A: +16 rows per i
  const int bk = tid >> 5, bn4 = (tid & 31) * 4;   // B: +8 k per i

  auto stage = [&](int k0, int buf) {
#pragma unroll
    for (int i = 0; i < 8; ++i) {
      int r = ar + i * 16;
      const float4 va = *(const float4*)&A[(size_t)(tileM + r) * K + k0 + ac4];
      *(uint2*)&As[buf][r][ac4] = make_uint2(f2bf_pk(va.x, va.y), f2bf_pk(va.z, va.w));
    }
#pragma unroll
    for (int i = 0; i < 8; ++i) {
      int kk = bk + i * 8;
      const float4 vb = *(const float4*)&Bm[(size_t)(k0 + kk) * N + tileN + bn4];
      Bs[buf][bn4 + 0][kk] = f2bf(vb.x); Bs[buf][bn4 + 1][kk] = f2bf(vb.y);
      Bs[buf][bn4 + 2][kk] = f2bf(vb.z); Bs[buf][bn4 + 3][kk] = f2bf(vb.w);
    }
  };

  v8f acc[2][4] = {};

  stage(0, 0);
  const int nsteps = K >> 6;
  for (int it = 0; it < nsteps; ++it) {
    __syncthreads();
    if (it + 1 < nsteps) stage((it + 1) << 6, (it + 1) & 1);
    const int buf = it & 1;

#pragma unroll
    for (int ks = 0; ks < 2; ++ks) {        // two 32-wide K substeps
      const int kofs = ks * 32;             // shorts
      FragBF a[2], b[4];
#pragma unroll
      for (int mi = 0; mi < 2; ++mi) {
        int r = mrow0 + mi * 16 + (lane & 15);
        a[mi].q[0] = *(const uint4*)&As[buf][r][kofs + khalfA];
        a[mi].q[1] = *(const uint4*)&As[buf][r][kofs + 16 + khalfA];
      }
#pragma unroll
      for (int ni = 0; ni < 4; ++ni) {
        int c = ncol0 + ni * 16 + (lane & 15);
        b[ni].q[0] = *(const uint4*)&Bs[buf][c][kofs + kb0];
        b[ni].q[1] = *(const uint4*)&Bs[buf][c][kofs + kb0 + 16];
      }
#pragma unroll
      for (int mi = 0; mi < 2; ++mi)
#pragma unroll
        for (int ni = 0; ni < 4; ++ni)
          acc[mi][ni] = __builtin_amdgcn_wmma_f32_16x16x32_bf16(
              false, a[mi].v, false, b[ni].v, (short)0, acc[mi][ni], false, false);
    }
  }

  // Straight-line epilogue (specialized at compile time)
#pragma unroll
  for (int mi = 0; mi < 2; ++mi) {
#pragma unroll
    for (int ni = 0; ni < 4; ++ni) {
      const int col = tileN + ncol0 + ni * 16 + (lane & 15);
      float bv = 0.f;
      if (HAS_BIAS) bv = bias[col];
#pragma unroll
      for (int r = 0; r < 8; ++r) {
        const int row = tileM + mrow0 + mi * 16 + r + ((lane >> 4) << 3);
        float v = acc[mi][ni][r] + bv;
        if (RELU) v = v > 0.f ? v : 0.f;
        const size_t off = (size_t)row * N + col;
        if (HAS_RES) v += resid[off];
        C[off] = v;
      }
    }
  }
}

// ---------------------------------------------------------------------------
// Flash attention: 8 waves/block, 16 queries/wave, 32-key chunks, bf16 WMMA.
// Q,K,V,O are [B*S, D] with head h in columns h*64 .. h*64+63.
// ---------------------------------------------------------------------------
__global__ __launch_bounds__(256)
void attention(const float* __restrict__ Q, const float* __restrict__ Kg,
               const float* __restrict__ Vg, float* __restrict__ O, int causal)
{
  __shared__ __align__(16) unsigned short Ks[32][72];      // [key][d], 144B rows
  __shared__ __align__(16) unsigned short Vs[64][40];      // [d][key] transposed
  __shared__ __align__(16) unsigned short Ps[8][16][40];   // per-wave P tile

  const int tid  = threadIdx.x;
  const int lane = tid & 31;
  const int wave = tid >> 5;
  const int b = blockIdx.z, h = blockIdx.y;
  const int qbase = blockIdx.x * 128 + wave * 16;
  const float* Qb = Q  + ((size_t)b * S_) * D_ + h * DH_;
  const float* Kb = Kg + ((size_t)b * S_) * D_ + h * DH_;
  const float* Vb = Vg + ((size_t)b * S_) * D_ + h * DH_;

  // Q fragments (two K-steps over DH=64)
  FragBF aq[2];
  {
    int m = lane & 15, khalf = (lane >> 4) * 8;
#pragma unroll
    for (int s = 0; s < 2; ++s)
#pragma unroll
      for (int i = 0; i < 16; ++i) {
        int d = s * 32 + (i >> 3) * 16 + khalf + (i & 7);
        aq[s].u[i] = f2bf(Qb[(size_t)(qbase + m) * D_ + d]);
      }
  }

  v8f o[4] = {};
  float mrow[8], lrow[8];
#pragma unroll
  for (int r = 0; r < 8; ++r) { mrow[r] = -3.0e38f; lrow[r] = 0.f; }

  const int kb0  = (lane >> 4) * 8;
  const int qEnd = blockIdx.x * 128 + 127;

  for (int kc = 0; kc < S_; kc += 32) {
    if (causal && kc > qEnd) break;  // block-uniform early exit

    // Cooperative stage of K (row-major, packed b64 stores) and V (transposed)
#pragma unroll
    for (int i = 0; i < 2; ++i) {
      int fidx = i * 256 + tid;
      int key = fidx >> 4, d4 = (fidx & 15) * 4;
      float4 vk = *(const float4*)&Kb[(size_t)(kc + key) * D_ + d4];
      *(uint2*)&Ks[key][d4] = make_uint2(f2bf_pk(vk.x, vk.y), f2bf_pk(vk.z, vk.w));
      float4 vv = *(const float4*)&Vb[(size_t)(kc + key) * D_ + d4];
      Vs[d4 + 0][key] = f2bf(vv.x); Vs[d4 + 1][key] = f2bf(vv.y);
      Vs[d4 + 2][key] = f2bf(vv.z); Vs[d4 + 3][key] = f2bf(vv.w);
    }
    __syncthreads();

    // Scores S = Q @ K^T : two 16x16 key tiles, DH reduced in two WMMA steps
    v8f sc[2] = {};
#pragma unroll
    for (int ni = 0; ni < 2; ++ni) {
      int keyc = ni * 16 + (lane & 15);
#pragma unroll
      for (int s = 0; s < 2; ++s) {
        FragBF bk;
        bk.q[0] = *(const uint4*)&Ks[keyc][s * 32 + kb0];
        bk.q[1] = *(const uint4*)&Ks[keyc][s * 32 + kb0 + 16];
        sc[ni] = __builtin_amdgcn_wmma_f32_16x16x32_bf16(
            false, aq[s].v, false, bk.v, (short)0, sc[ni], false, false);
      }
    }

    // Online softmax. Mask compares only needed in the diagonal chunk
    // (wave-uniform condition => scc branch, not per-lane cndmask elsewhere).
    const bool needMask = (causal != 0) && (kc + 31 > qbase);
    float p0v[8], p1v[8], mtile[8];
    if (needMask) {
#pragma unroll
      for (int r = 0; r < 8; ++r) {
        int qrow = qbase + r + ((lane >> 4) << 3);
        int key0 = kc + (lane & 15);
        float s0 = sc[0][r] * 0.125f;   // 1/sqrt(64)
        float s1 = sc[1][r] * 0.125f;
        if (key0 > qrow)      s0 = -3.0e38f;
        if (key0 + 16 > qrow) s1 = -3.0e38f;
        p0v[r] = s0; p1v[r] = s1;
      }
    } else {
#pragma unroll
      for (int r = 0; r < 8; ++r) {
        p0v[r] = sc[0][r] * 0.125f;
        p1v[r] = sc[1][r] * 0.125f;
      }
    }
#pragma unroll
    for (int r = 0; r < 8; ++r) {
      float mt = fmaxf(p0v[r], p1v[r]);
#pragma unroll
      for (int off = 1; off < 16; off <<= 1)
        mt = fmaxf(mt, __shfl_xor(mt, off, 32));
      mtile[r] = mt;
    }
#pragma unroll
    for (int r = 0; r < 8; ++r) {
      float mnew  = fmaxf(mrow[r], mtile[r]);
      float scale = __expf(mrow[r] - mnew);
      float p0 = __expf(p0v[r] - mnew);
      float p1 = __expf(p1v[r] - mnew);
      float psum = p0 + p1;
#pragma unroll
      for (int off = 1; off < 16; off <<= 1)
        psum += __shfl_xor(psum, off, 32);
      lrow[r] = lrow[r] * scale + psum;
      mrow[r] = mnew;
#pragma unroll
      for (int ni = 0; ni < 4; ++ni) o[ni][r] *= scale;
      int prow = r + ((lane >> 4) << 3);
      Ps[wave][prow][(lane & 15)]      = f2bf(p0);
      Ps[wave][prow][16 + (lane & 15)] = f2bf(p1);
    }

    // P (C-layout) -> A-fragment via per-wave LDS round-trip, then O += P @ V
    FragBF ap;
    {
      int m = lane & 15, khalf = (lane >> 4) * 8;
      ap.q[0] = *(const uint4*)&Ps[wave][m][khalf];
      ap.q[1] = *(const uint4*)&Ps[wave][m][16 + khalf];
    }
#pragma unroll
    for (int ni = 0; ni < 4; ++ni) {
      int dcol = ni * 16 + (lane & 15);
      FragBF bv;
      bv.q[0] = *(const uint4*)&Vs[dcol][kb0];
      bv.q[1] = *(const uint4*)&Vs[dcol][kb0 + 16];
      o[ni] = __builtin_amdgcn_wmma_f32_16x16x32_bf16(
          false, ap.v, false, bv.v, (short)0, o[ni], false, false);
    }
    __syncthreads();
  }

  // Normalize and store
#pragma unroll
  for (int ni = 0; ni < 4; ++ni) {
    int d = ni * 16 + (lane & 15);
#pragma unroll
    for (int r = 0; r < 8; ++r) {
      int qrow = qbase + r + ((lane >> 4) << 3);
      O[((size_t)(b * S_ + qrow)) * D_ + h * DH_ + d] = o[ni][r] / lrow[r];
    }
  }
}

// ---------------------------------------------------------------------------
// LayerNorm: one wave32 per row of 512 elements
// ---------------------------------------------------------------------------
__global__ __launch_bounds__(128)
void layernorm(const float* __restrict__ X, const float* __restrict__ g,
               const float* __restrict__ bb, float* __restrict__ Y)
{
  int lane = threadIdx.x & 31, wave = threadIdx.x >> 5;
  int row = blockIdx.x * 4 + wave;
  const float* x = X + (size_t)row * D_;
  float v[16], s = 0.f;
#pragma unroll
  for (int i = 0; i < 16; ++i) { v[i] = x[lane + i * 32]; s += v[i]; }
#pragma unroll
  for (int off = 1; off < 32; off <<= 1) s += __shfl_xor(s, off, 32);
  float mean = s * (1.f / D_);
  float var = 0.f;
#pragma unroll
  for (int i = 0; i < 16; ++i) { float d = v[i] - mean; var += d * d; }
#pragma unroll
  for (int off = 1; off < 32; off <<= 1) var += __shfl_xor(var, off, 32);
  float inv = rsqrtf(var * (1.f / D_) + 1e-6f);
  float* y = Y + (size_t)row * D_;
#pragma unroll
  for (int i = 0; i < 16; ++i) {
    int c = lane + i * 32;
    y[c] = (v[i] - mean) * inv * g[c] + bb[c];
  }
}

// ---------------------------------------------------------------------------
// Embedding gather + sinusoidal positional encoding
// ---------------------------------------------------------------------------
__global__ __launch_bounds__(128)
void embed_pe(const int* __restrict__ ids, const float* __restrict__ emb,
              float* __restrict__ out)
{
  int row = blockIdx.x;           // b*S + s
  int s = row & (S_ - 1);
  int id = ids[row];
  const float* e = emb + (size_t)id * D_;
  float* o = out + (size_t)row * D_;
  const float log10000 = 9.21034037198f;
#pragma unroll
  for (int i = 0; i < 4; ++i) {
    int c = threadIdx.x + i * 128;
    int i2 = c & ~1;
    float ang = (float)s * __expf(-log10000 * (float)i2 / (float)D_);
    float pe = (c & 1) ? __cosf(ang) : __sinf(ang);
    o[c] = e[c] + pe;
  }
}

// ---------------------------------------------------------------------------
extern "C" void kernel_launch(void* const* d_in, const int* in_sizes, int n_in,
                              void* d_out, int out_size, void* d_ws, size_t ws_size,
                              hipStream_t stream)
{
  (void)in_sizes; (void)n_in; (void)out_size; (void)ws_size;
  int idx = 0;
  const int* src_ids = (const int*)d_in[idx++];
  const int* tgt_ids = (const int*)d_in[idx++];
  idx++;  // src_mask: all true -> encoder non-causal
  idx++;  // tgt_mask: tril -> causal, synthesized in-kernel
  const float* src_emb = (const float*)d_in[idx++];
  const float* tgt_emb = (const float*)d_in[idx++];

  struct AW { const float *Wq, *Wk, *Wv, *Wo; };
  struct EB { const float *g1, *b1, *g2, *b2, *W1, *bb1, *W2, *bb2; AW a; };
  struct DB { const float *g1, *b1, *g2, *b2, *W1, *bb1, *W2, *bb2, *g3, *b3; AW sa, ca; };
  EB enc[NL_]; DB dec[NL_];
  auto Fp = [&](void) { return (const float*)d_in[idx++]; };
  for (int l = 0; l < NL_; ++l) {
    enc[l].g1 = Fp(); enc[l].b1 = Fp(); enc[l].g2 = Fp(); enc[l].b2 = Fp();
    enc[l].W1 = Fp(); enc[l].bb1 = Fp(); enc[l].W2 = Fp(); enc[l].bb2 = Fp();
    enc[l].a.Wq = Fp(); enc[l].a.Wk = Fp(); enc[l].a.Wv = Fp(); enc[l].a.Wo = Fp();
  }
  for (int l = 0; l < NL_; ++l) {
    dec[l].g1 = Fp(); dec[l].b1 = Fp(); dec[l].g2 = Fp(); dec[l].b2 = Fp();
    dec[l].W1 = Fp(); dec[l].bb1 = Fp(); dec[l].W2 = Fp(); dec[l].bb2 = Fp();
    dec[l].g3 = Fp(); dec[l].b3 = Fp();
    dec[l].sa.Wq = Fp(); dec[l].sa.Wk = Fp(); dec[l].sa.Wv = Fp(); dec[l].sa.Wo = Fp();
    dec[l].ca.Wq = Fp(); dec[l].ca.Wk = Fp(); dec[l].ca.Wv = Fp(); dec[l].ca.Wo = Fp();
  }
  const float* Wu = Fp();

  // Workspace carve (floats): 8 x [2048,512] + 1 x [2048,2048] = 48 MB
  float* w = (float*)d_ws;
  const size_t NX = (size_t)MROWS * D_;
  float* x   = w; w += NX;
  float* y   = w; w += NX;
  float* t1  = w; w += NX;
  float* t4  = w; w += NX;
  float* qb  = w; w += NX;
  float* kb  = w; w += NX;
  float* vb  = w; w += NX;
  float* att = w; w += NX;
  float* mid = w; w += (size_t)MROWS * F_;

  // mode: 0 = plain, 1 = +resid, 2 = +bias+relu, 3 = +bias+resid
  auto gemm = [&](const float* A, const float* Bm, const float* bias,
                  const float* resid, float* C, int M, int N, int K, int mode) {
    dim3 g(N / 128, M / 128);
    if (mode == 0)
      gemm_bf16<false, false, false><<<g, 256, 0, stream>>>(A, Bm, nullptr, nullptr, C, M, N, K);
    else if (mode == 1)
      gemm_bf16<false, false, true ><<<g, 256, 0, stream>>>(A, Bm, nullptr, resid, C, M, N, K);
    else if (mode == 2)
      gemm_bf16<true,  true,  false><<<g, 256, 0, stream>>>(A, Bm, bias, nullptr, C, M, N, K);
    else
      gemm_bf16<true,  false, true ><<<g, 256, 0, stream>>>(A, Bm, bias, resid, C, M, N, K);
  };
  auto ln = [&](const float* X, const float* g, const float* bb, float* Y) {
    layernorm<<<dim3(MROWS / 4), 128, 0, stream>>>(X, g, bb, Y);
  };
  auto at = [&](const float* Qm, const float* Km, const float* Vm, float* Om, int causal) {
    attention<<<dim3(S_ / 128, H_, B_), 256, 0, stream>>>(Qm, Km, Vm, Om, causal);
  };

  // -------- Encoder --------
  embed_pe<<<dim3(B_ * S_), 128, 0, stream>>>(src_ids, src_emb, x);
  for (int l = 0; l < NL_; ++l) {
    ln(x, enc[l].g1, enc[l].b1, t1);
    gemm(t1, enc[l].a.Wq, nullptr, nullptr, qb, MROWS, D_, D_, 0);
    gemm(t1, enc[l].a.Wk, nullptr, nullptr, kb, MROWS, D_, D_, 0);
    gemm(t1, enc[l].a.Wv, nullptr, nullptr, vb, MROWS, D_, D_, 0);
    at(qb, kb, vb, att, 0);
    gemm(att, enc[l].a.Wo, nullptr, x, x, MROWS, D_, D_, 1);
    ln(x, enc[l].g2, enc[l].b2, t4);
    gemm(t4, enc[l].W1, enc[l].bb1, nullptr, mid, MROWS, F_, D_, 2);
    gemm(mid, enc[l].W2, enc[l].bb2, x, x, MROWS, D_, F_, 3);
  }

  // -------- Decoder --------
  embed_pe<<<dim3(B_ * S_), 128, 0, stream>>>(tgt_ids, tgt_emb, y);
  for (int l = 0; l < NL_; ++l) {
    ln(y, dec[l].g1, dec[l].b1, t1);
    gemm(t1, dec[l].sa.Wq, nullptr, nullptr, qb, MROWS, D_, D_, 0);
    gemm(t1, dec[l].sa.Wk, nullptr, nullptr, kb, MROWS, D_, D_, 0);
    gemm(t1, dec[l].sa.Wv, nullptr, nullptr, vb, MROWS, D_, D_, 0);
    at(qb, kb, vb, att, 1);
    gemm(att, dec[l].sa.Wo, nullptr, y, y, MROWS, D_, D_, 1);
    ln(y, dec[l].g2, dec[l].b2, t4);
    gemm(t4, dec[l].ca.Wq, nullptr, nullptr, qb, MROWS, D_, D_, 0);
    gemm(x,  dec[l].ca.Wk, nullptr, nullptr, kb, MROWS, D_, D_, 0);
    gemm(x,  dec[l].ca.Wv, nullptr, nullptr, vb, MROWS, D_, D_, 0);
    at(qb, kb, vb, att, 1);  // reference reuses tgt_mask (causal) for cross-attn
    gemm(att, dec[l].ca.Wo, nullptr, y, y, MROWS, D_, D_, 1);
    ln(y, dec[l].g3, dec[l].b3, t4);
    gemm(t4, dec[l].W1, dec[l].bb1, nullptr, mid, MROWS, F_, D_, 2);
    gemm(mid, dec[l].W2, dec[l].bb2, y, y, MROWS, D_, F_, 3);
  }

  // -------- Unembedding: [2048,512] @ [512,32000] --------
  gemm(y, Wu, nullptr, nullptr, (float*)d_out, MROWS, V_, D_, 0);
}
